// GenericLossFirstPart_16226386444969
// MI455X (gfx1250) — matrix operand order repaired and checked by reading, compiled, not run on hardware
//
#include <hip/hip_runtime.h>
#include <hip/hip_bf16.h>

// Problem constants (match reference)
#define BATCH 16384
#define DIM   2048
#define NCLS  1000
#define EPSV  1e-12f

#define NMT  (BATCH / 16)   // 1024 m-tiles
#define NNTP 64             // padded n-tiles (64*16 = 1024 >= 1000)
#define NKT  (DIM / 32)     // 64 k-steps of 32

// Fast-path GEMM tiling: 128x128 block, 8 waves, 32x64 per wave
#define BM 128
#define BN 128

typedef __attribute__((ext_vector_type(16))) __bf16 v16bf;
typedef __attribute__((ext_vector_type(8)))  float  v8f;

union Frag { uint4 q[2]; v16bf v; };

__device__ __forceinline__ unsigned bcu(float f) { return __builtin_bit_cast(unsigned, f); }
__device__ __forceinline__ float    bcf(unsigned u) { return __builtin_bit_cast(float, u); }

// fp32 -> packed bf16 hi (truncation) + bf16 lo (residual)
__device__ __forceinline__ void split4(const float4 f, uint2& h, uint2& l) {
    unsigned hx = bcu(f.x) & 0xFFFF0000u;
    unsigned hy = bcu(f.y) & 0xFFFF0000u;
    unsigned hz = bcu(f.z) & 0xFFFF0000u;
    unsigned hw = bcu(f.w) & 0xFFFF0000u;
    h.x = (hx >> 16) | hy;
    h.y = (hz >> 16) | hw;
    float rx = f.x - bcf(hx);
    float ry = f.y - bcf(hy);
    float rz = f.z - bcf(hz);
    float rw = f.w - bcf(hw);
    l.x = (bcu(rx) >> 16) | (bcu(ry) & 0xFFFF0000u);
    l.y = (bcu(rz) >> 16) | (bcu(rw) & 0xFFFF0000u);
}

__device__ __forceinline__ void split8(const float4 a, const float4 b, uint4& h, uint4& l) {
    uint2 ha, la, hb, lb;
    split4(a, ha, la);
    split4(b, hb, lb);
    h.x = ha.x; h.y = ha.y; h.z = hb.x; h.w = hb.y;
    l.x = la.x; l.y = la.y; l.z = lb.x; l.w = lb.y;
}

// -------------------- squared row norms (one wave per row) --------------------
__global__ void row_sqnorm_kernel(const float* __restrict__ src,
                                  float* __restrict__ dst,
                                  int nrows, int ncols) {
    const int wave = threadIdx.x >> 5;
    const int lane = threadIdx.x & 31;
    const int row  = blockIdx.x * 8 + wave;
    if (row >= nrows) return;
    const float* p = src + (size_t)row * ncols;
    float s = 0.0f;
    for (int c = lane * 4; c < ncols; c += 32 * 4) {
        float4 v = *(const float4*)(p + c);
        s += v.x * v.x + v.y * v.y + v.z * v.z + v.w * v.w;
    }
    #pragma unroll
    for (int off = 16; off >= 1; off >>= 1)
        s += __shfl_xor(s, off, 32);
    if (lane == 0) dst[row] = s;
}

// -------------------- one-time fragment-order swizzle of x --------------------
// Tile (mtg, kt) stored at uint4 index (kt*NMT + mtg)*64:
//   [0,32)  = each lane's q0 (A-layout: K = half*8..+7)
//   [32,64) = each lane's q1 (A-layout: K = 16+half*8..+7)
__global__ void __launch_bounds__(256)
xswz_kernel(const float* __restrict__ x, uint4* __restrict__ hi, uint4* __restrict__ lo) {
    const int wave = threadIdx.x >> 5;
    const int lane = threadIdx.x & 31;
    const int tile = blockIdx.x * 8 + wave;        // tile = kt*NMT + mtg
    const int mtg  = tile & (NMT - 1);
    const int kt   = tile >> 10;
    const int half = lane >> 4;
    const int lr   = lane & 15;

    const float* p = x + (size_t)(mtg * 16 + lr) * DIM + kt * 32;
    float4 a0 = *(const float4*)(p + half * 8);
    float4 a1 = *(const float4*)(p + half * 8 + 4);
    float4 b0 = *(const float4*)(p + 16 + half * 8);
    float4 b1 = *(const float4*)(p + 16 + half * 8 + 4);

    uint4 h0, l0, h1, l1;
    split8(a0, a1, h0, l0);
    split8(b0, b1, h1, l1);

    const size_t base = (size_t)tile * 64;
    hi[base + lane] = h0; hi[base + 32 + lane] = h1;
    lo[base + lane] = l0; lo[base + 32 + lane] = l1;
}

// -------------------- one-time fragment-order swizzle of w --------------------
// Tile (ntg, kt) at uint4 index (kt*NNTP + ntg)*64. B-layout: lane = column,
// q0 = K[half*8..+7], q1 = K[16+half*8..+7]. Columns >= NCLS zero-padded.
__global__ void __launch_bounds__(256)
wswz_kernel(const float* __restrict__ w, uint4* __restrict__ hi, uint4* __restrict__ lo) {
    const int wave = threadIdx.x >> 5;
    const int lane = threadIdx.x & 31;
    const int tile = blockIdx.x * 8 + wave;        // tile = kt*NNTP + ntg
    const int ntg  = tile & (NNTP - 1);
    const int kt   = tile >> 6;
    const int half = lane >> 4;
    const int lr   = lane & 15;
    const int c    = ntg * 16 + lr;

    float4 a0 = {0.f, 0.f, 0.f, 0.f}, a1 = a0, b0 = a0, b1 = a0;
    if (c < NCLS) {
        const float* p = w + (size_t)c * DIM + kt * 32 + half * 8;
        a0 = *(const float4*)(p);
        a1 = *(const float4*)(p + 4);
        b0 = *(const float4*)(p + 16);
        b1 = *(const float4*)(p + 20);
    }
    uint4 h0, l0, h1, l1;
    split8(a0, a1, h0, l0);
    split8(b0, b1, h1, l1);

    const size_t base = (size_t)tile * 64;
    hi[base + lane] = h0; hi[base + 32 + lane] = h1;
    lo[base + lane] = l0; lo[base + 32 + lane] = l1;
}

// -------------------- fast path: pure-bf16 WMMA distance GEMM --------------------
__global__ void __launch_bounds__(256)
dml_wmma_fast_kernel(const uint4* __restrict__ xhi, const uint4* __restrict__ xlo,
                     const uint4* __restrict__ whi, const uint4* __restrict__ wlo,
                     const float* __restrict__ scales,
                     const float* __restrict__ x2, const float* __restrict__ w2,
                     float* __restrict__ out) {
    // 8 tiles per operand per k-step, fragment-ordered; double buffered = 64 KB
    __shared__ uint4 sAh[2][512];
    __shared__ uint4 sAl[2][512];
    __shared__ uint4 sBh[2][512];
    __shared__ uint4 sBl[2][512];

    const int t    = threadIdx.x;
    const int lane = t & 31;
    const int wave = t >> 5;
    const int wm   = wave >> 1;    // 0..3 -> 32 rows each
    const int wn   = wave & 1;     // 0..1 -> 64 cols each
    const int half = lane >> 4;
    const int lr   = lane & 15;

    const int row0 = blockIdx.x * BM;
    const int c0   = blockIdx.y * BN;
    const size_t aOff = (size_t)blockIdx.x * 8 * 64;   // uint4 offset of this block's m-tiles
    const size_t bOff = (size_t)blockIdx.y * 8 * 64;   // uint4 offset of this block's n-tiles

    v8f acc[2][4];
    #pragma unroll
    for (int mt = 0; mt < 2; ++mt)
        #pragma unroll
        for (int nt = 0; nt < 4; ++nt)
            acc[mt][nt] = (v8f){0.f, 0.f, 0.f, 0.f, 0.f, 0.f, 0.f, 0.f};

    // prologue: stage k-step 0
    #pragma unroll
    for (int i = 0; i < 2; ++i) {
        const int j = t + i * 256;
        sAh[0][j] = xhi[aOff + j];
        sAl[0][j] = xlo[aOff + j];
        sBh[0][j] = whi[bOff + j];
        sBl[0][j] = wlo[bOff + j];
    }
    __syncthreads();

    for (int ks = 0; ks < NKT; ++ks) {
        const int cur = ks & 1;
        const bool more = (ks + 1) < NKT;

        // issue next k-step's global loads early (latency hidden behind WMMAs)
        uint4 tAh[2], tAl[2], tBh[2], tBl[2];
        if (more) {
            const size_t ga = (size_t)(ks + 1) * (NMT * 64) + aOff;
            const size_t gb = (size_t)(ks + 1) * (NNTP * 64) + bOff;
            #pragma unroll
            for (int i = 0; i < 2; ++i) {
                const int j = t + i * 256;
                tAh[i] = xhi[ga + j];
                tAl[i] = xlo[ga + j];
                tBh[i] = whi[gb + j];
                tBl[i] = wlo[gb + j];
            }
        }

        // A fragments for this wave's two m-tiles (hi+lo)
        Frag ah[2], al[2];
        #pragma unroll
        for (int mt = 0; mt < 2; ++mt) {
            const int ta = (wm * 2 + mt) * 64 + lane;
            ah[mt].q[0] = sAh[cur][ta];
            ah[mt].q[1] = sAh[cur][ta + 32];
            al[mt].q[0] = sAl[cur][ta];
            al[mt].q[1] = sAl[cur][ta + 32];
        }

        // per n-tile: load B frags, 6 WMMAs (hi*hi + hi*lo + lo*hi for 2 m-tiles)
        #pragma unroll
        for (int nt = 0; nt < 4; ++nt) {
            const int tb = (wn * 4 + nt) * 64 + lane;
            Frag bh, bl;
            bh.q[0] = sBh[cur][tb];
            bh.q[1] = sBh[cur][tb + 32];
            bl.q[0] = sBl[cur][tb];
            bl.q[1] = sBl[cur][tb + 32];
            #pragma unroll
            for (int mt = 0; mt < 2; ++mt) {
                acc[mt][nt] = __builtin_amdgcn_wmma_f32_16x16x32_bf16(
                    false, ah[mt].v, false, bh.v, (short)0, acc[mt][nt], false, false);
                acc[mt][nt] = __builtin_amdgcn_wmma_f32_16x16x32_bf16(
                    false, ah[mt].v, false, bl.v, (short)0, acc[mt][nt], false, false);
                acc[mt][nt] = __builtin_amdgcn_wmma_f32_16x16x32_bf16(
                    false, al[mt].v, false, bh.v, (short)0, acc[mt][nt], false, false);
            }
        }

        // commit next tile to the other LDS buffer
        if (more) {
            #pragma unroll
            for (int i = 0; i < 2; ++i) {
                const int j = t + i * 256;
                sAh[cur ^ 1][j] = tAh[i];
                sAl[cur ^ 1][j] = tAl[i];
                sBh[cur ^ 1][j] = tBh[i];
                sBl[cur ^ 1][j] = tBl[i];
            }
        }
        __syncthreads();
    }

    // epilogue: d2 = x2 + w2 - 2*cross ; out = -scale * sqrt(max(d2, eps))
    const float scale = *scales;
    float xr[2][8];
    #pragma unroll
    for (int mt = 0; mt < 2; ++mt)
        #pragma unroll
        for (int v = 0; v < 8; ++v)
            xr[mt][v] = x2[row0 + wm * 32 + mt * 16 + half * 8 + v];

    #pragma unroll
    for (int nt = 0; nt < 4; ++nt) {
        const int col = c0 + wn * 64 + nt * 16 + lr;   // lane lr -> N per C/D layout
        if (col < NCLS) {
            const float w2v = w2[col];
            #pragma unroll
            for (int mt = 0; mt < 2; ++mt) {
                #pragma unroll
                for (int v = 0; v < 8; ++v) {          // VGPR v -> M = v + 8*half
                    const int row = row0 + wm * 32 + mt * 16 + half * 8 + v;
                    float d2 = xr[mt][v] + w2v - 2.0f * acc[mt][nt][v];
                    d2 = fmaxf(d2, EPSV);
                    out[(size_t)row * NCLS + col] = -scale * sqrtf(d2);
                }
            }
        }
    }
}

// -------------------- fallback: in-loop conversion (small workspace) --------------------
#define FBN 64
#define LDSROW 20
__global__ void __launch_bounds__(256)
dml_wmma_fallback_kernel(const float* __restrict__ x,
                         const float* __restrict__ w,
                         const float* __restrict__ scales,
                         const float* __restrict__ x2,
                         const float* __restrict__ w2,
                         float* __restrict__ out) {
    __shared__ __align__(16) unsigned sAh[128 * LDSROW];
    __shared__ __align__(16) unsigned sAl[128 * LDSROW];
    __shared__ __align__(16) unsigned sBh[FBN * LDSROW];
    __shared__ __align__(16) unsigned sBl[FBN * LDSROW];

    const int t    = threadIdx.x;
    const int lane = t & 31;
    const int wave = t >> 5;
    const int wm   = wave >> 1;
    const int wn   = wave & 1;
    const int half = lane >> 4;
    const int lr   = lane & 15;

    const int row0 = blockIdx.x * 128;
    const int c0   = blockIdx.y * FBN;

    v8f acc[2][2];
    #pragma unroll
    for (int mt = 0; mt < 2; ++mt)
        #pragma unroll
        for (int nt = 0; nt < 2; ++nt)
            acc[mt][nt] = (v8f){0.f, 0.f, 0.f, 0.f, 0.f, 0.f, 0.f, 0.f};

    for (int k0 = 0; k0 < DIM; k0 += 32) {
        #pragma unroll
        for (int i = 0; i < 4; ++i) {
            const int id  = t + i * 256;
            const int row = id >> 3;
            const int kc  = id & 7;
            const float4 f = *(const float4*)(x + (size_t)(row0 + row) * DIM + k0 + kc * 4);
            uint2 h, l;
            split4(f, h, l);
            const int off = row * LDSROW + kc * 2;
            *(uint2*)(sAh + off) = h;
            *(uint2*)(sAl + off) = l;
        }
        #pragma unroll
        for (int i = 0; i < 2; ++i) {
            const int id  = t + i * 256;
            const int col = id >> 3;
            const int kc  = id & 7;
            const int c   = c0 + col;
            float4 f = {0.f, 0.f, 0.f, 0.f};
            if (c < NCLS)
                f = *(const float4*)(w + (size_t)c * DIM + k0 + kc * 4);
            uint2 h, l;
            split4(f, h, l);
            const int off = col * LDSROW + kc * 2;
            *(uint2*)(sBh + off) = h;
            *(uint2*)(sBl + off) = l;
        }
        __syncthreads();

        Frag ah[2], al[2], bh[2], bl[2];
        #pragma unroll
        for (int mt = 0; mt < 2; ++mt) {
            const int arow = wm * 32 + mt * 16 + lr;
            const char* ph = (const char*)sAh + arow * 80 + half * 16;
            const char* pl = (const char*)sAl + arow * 80 + half * 16;
            ah[mt].q[0] = *(const uint4*)(ph);
            ah[mt].q[1] = *(const uint4*)(ph + 32);
            al[mt].q[0] = *(const uint4*)(pl);
            al[mt].q[1] = *(const uint4*)(pl + 32);
        }
        #pragma unroll
        for (int nt = 0; nt < 2; ++nt) {
            const int bcol = wn * 32 + nt * 16 + lr;
            const char* ph = (const char*)sBh + bcol * 80 + half * 16;
            const char* pl = (const char*)sBl + bcol * 80 + half * 16;
            bh[nt].q[0] = *(const uint4*)(ph);
            bh[nt].q[1] = *(const uint4*)(ph + 32);
            bl[nt].q[0] = *(const uint4*)(pl);
            bl[nt].q[1] = *(const uint4*)(pl + 32);
        }

        #pragma unroll
        for (int mt = 0; mt < 2; ++mt) {
            #pragma unroll
            for (int nt = 0; nt < 2; ++nt) {
                acc[mt][nt] = __builtin_amdgcn_wmma_f32_16x16x32_bf16(
                    false, ah[mt].v, false, bh[nt].v, (short)0, acc[mt][nt], false, false);
                acc[mt][nt] = __builtin_amdgcn_wmma_f32_16x16x32_bf16(
                    false, ah[mt].v, false, bl[nt].v, (short)0, acc[mt][nt], false, false);
                acc[mt][nt] = __builtin_amdgcn_wmma_f32_16x16x32_bf16(
                    false, al[mt].v, false, bh[nt].v, (short)0, acc[mt][nt], false, false);
            }
        }
        __syncthreads();
    }

    const float scale = *scales;
    float xr[2][8];
    #pragma unroll
    for (int mt = 0; mt < 2; ++mt)
        #pragma unroll
        for (int v = 0; v < 8; ++v)
            xr[mt][v] = x2[row0 + wm * 32 + mt * 16 + half * 8 + v];

    #pragma unroll
    for (int nt = 0; nt < 2; ++nt) {
        const int col = c0 + wn * 32 + nt * 16 + lr;
        if (col < NCLS) {
            const float w2v = w2[col];
            #pragma unroll
            for (int mt = 0; mt < 2; ++mt) {
                #pragma unroll
                for (int v = 0; v < 8; ++v) {
                    const int row = row0 + wm * 32 + mt * 16 + half * 8 + v;
                    float d2 = xr[mt][v] + w2v - 2.0f * acc[mt][nt][v];
                    d2 = fmaxf(d2, EPSV);
                    out[(size_t)row * NCLS + col] = -scale * sqrtf(d2);
                }
            }
        }
    }
}

extern "C" void kernel_launch(void* const* d_in, const int* in_sizes, int n_in,
                              void* d_out, int out_size, void* d_ws, size_t ws_size,
                              hipStream_t stream) {
    const float* x  = (const float*)d_in[0];   // [BATCH, DIM]
    const float* w  = (const float*)d_in[1];   // [NCLS, DIM]
    const float* sc = (const float*)d_in[2];   // [1]
    float* out = (float*)d_out;                // [BATCH, NCLS]

    float* x2 = (float*)d_ws;                  // [BATCH]
    float* w2 = x2 + BATCH;                    // [NCLS]

    // 1) squared norms (deterministic wave reductions)
    row_sqnorm_kernel<<<dim3((BATCH + 7) / 8), dim3(256), 0, stream>>>(x, x2, BATCH, DIM);
    row_sqnorm_kernel<<<dim3((NCLS + 7) / 8),  dim3(256), 0, stream>>>(w, w2, NCLS, DIM);

    const size_t normBytes = ((size_t)(BATCH + NCLS) * sizeof(float) + 255) & ~(size_t)255;
    const size_t xBytes    = (size_t)BATCH * DIM * 2;        // bf16 plane of x
    const size_t wBytes    = (size_t)NNTP * 16 * DIM * 2;    // bf16 plane of padded w
    const size_t need      = normBytes + 2 * xBytes + 2 * wBytes;   // ~142.7 MB

    if (ws_size >= need) {
        char* base = (char*)d_ws + normBytes;
        uint4* xhi = (uint4*)(base);
        uint4* xlo = (uint4*)(base + xBytes);
        uint4* whi = (uint4*)(base + 2 * xBytes);
        uint4* wlo = (uint4*)(base + 2 * xBytes + wBytes);

        // 2) one-time fp32 -> split-bf16 fragment-order swizzle (L2-resident afterwards)
        xswz_kernel<<<dim3((NMT * NKT) / 8), dim3(256), 0, stream>>>(x, xhi, xlo);
        wswz_kernel<<<dim3((NNTP * NKT) / 8), dim3(256), 0, stream>>>(w, whi, wlo);

        // 3) conversion-free double-buffered WMMA GEMM + fused distance epilogue
        dml_wmma_fast_kernel<<<dim3(BATCH / BM, (NNTP * 16) / BN), dim3(256), 0, stream>>>(
            xhi, xlo, whi, wlo, sc, x2, w2, out);
    } else {
        dml_wmma_fallback_kernel<<<dim3(BATCH / 128, (NCLS + FBN - 1) / FBN), dim3(256), 0, stream>>>(
            x, w, sc, x2, w2, out);
    }
}
